// Encoder_30116310680397
// MI455X (gfx1250) — matrix operand order repaired and checked by reading, not compiled, not run on hardware
//
#include <hip/hip_runtime.h>
#include <hip/hip_bf16.h>

// ---------------------------------------------------------------------------
// Bidirectional GRU encoder for MI455X (gfx1250), wave32 + WMMA bf16.
//   V=32000, E=512, H=1024, L=128, B=64
// Strategy:
//   1. embed + convert to bf16                       (parallel)
//   2. transpose+convert weights to bf16 W^T         (parallel)
//   3. gx = x @ Wx + bx as one big WMMA GEMM         (parallel, per direction)
//      -- register double-buffered so WMMAs don't stall on their own loads
//   4. ONE persistent kernel per direction runs all 128 timesteps:
//        - recurrent weights preloaded to LDS (async global->LDS, 192KB/WGP)
//        - phase 1: gh = h@Whzr -> z, r, r*h   (128 waves, B-frags from LDS)
//        - phase 2: (r*h)@Whh -> hcand, h update (64 waves)
//        - phases separated by an agent-scope atomic grid barrier
// ---------------------------------------------------------------------------

typedef __bf16 bf16_t;
typedef __attribute__((ext_vector_type(16))) __bf16 v16bf;
typedef __attribute__((ext_vector_type(8)))  __bf16 v8bf;
typedef __attribute__((ext_vector_type(8)))  float  v8f;

constexpr int Lc = 128;    // sequence length
constexpr int Bc = 64;     // batch
constexpr int Ec = 512;    // embedding dim
constexpr int Hc = 1024;   // hidden dim
constexpr int G3 = 3 * Hc; // 3072 gate columns of Wx

constexpr int SCAN_BLOCKS  = 32;   // persistent grid
constexpr int SCAN_THREADS = 128;  // 4 wave32 waves / block

// ---------------- WMMA fragment helpers (CDNA5 ISA 7.12.2) ------------
// A-matrix 16x32 bf16, row-major source, leading dim ld (elements):
//   lanes 0-15 : row m=lane,    K=[0..15]
//   lanes 16-31: row m=lane-16, K=[16..31]
__device__ inline v16bf load_fragA(const bf16_t* base, int ld) {
  int lane = threadIdx.x & 31;
  int m    = lane & 15;
  int kb   = (lane >> 4) << 4;
  const bf16_t* p = base + (long)m * ld + kb;
  v8bf lo = *(const v8bf*)(p);
  v8bf hi = *(const v8bf*)(p + 8);
  v16bf r;
#pragma unroll
  for (int i = 0; i < 8; ++i) { r[i] = lo[i]; r[i + 8] = hi[i]; }
  return r;
}

// B-matrix 32x16 bf16 from N-major/K-contiguous (W^T) storage:
//   lanes 0-15 : col n=lane,    K=[0..15]  (one contiguous 32B read)
//   lanes 16-31: col n=lane-16, K=[16..31]
__device__ inline v16bf load_fragB(const bf16_t* baseT, int ld) {
  int lane = threadIdx.x & 31;
  int n    = lane & 15;
  int kb   = (lane >> 4) << 4;
  return *(const v16bf*)(baseT + (long)n * ld + kb);
}

__device__ inline v8f wmma_bf16(v16bf a, v16bf b, v8f c) {
  return __builtin_amdgcn_wmma_f32_16x16x32_bf16(
      false, a, false, b, (short)0, c, false, false);
}

__device__ inline float sigmoidf_(float x) { return 1.0f / (1.0f + __expf(-x)); }

// ---------------- grid-wide barrier (monotonic counter, agent scope) -------
__device__ inline void grid_barrier(unsigned* bar, unsigned target) {
  __syncthreads();
  __threadfence();
  if (threadIdx.x == 0) {
    __hip_atomic_fetch_add(bar, 1u, __ATOMIC_ACQ_REL, __HIP_MEMORY_SCOPE_AGENT);
    while (__hip_atomic_load(bar, __ATOMIC_ACQUIRE, __HIP_MEMORY_SCOPE_AGENT) < target) {
      __builtin_amdgcn_s_sleep(2);
    }
  }
  __syncthreads();
  __threadfence();
}

// ---------------- setup kernels -------------------------------------------
__global__ void embed_cvt_kernel(const int* __restrict__ xs,
                                 const float* __restrict__ emb,
                                 bf16_t* __restrict__ xe) {
  long i  = (long)blockIdx.x * 256 + threadIdx.x;    // over L*B*E
  int  e  = (int)(i & (Ec - 1));
  long tb = i >> 9;                                   // /512
  int tok = xs[tb];
  xe[i] = (bf16_t)emb[(long)tok * Ec + e];
}

// dst[n*K + k] = bf16(src[k*N + n])
__global__ void transpose_cvt_kernel(const float* __restrict__ src,
                                     bf16_t* __restrict__ dst, int K, int N) {
  long i = (long)blockIdx.x * 256 + threadIdx.x;
  if (i >= (long)K * N) return;
  int n = (int)(i % N);
  int k = (int)(i / N);
  dst[(long)n * K + k] = (bf16_t)src[i];
}

__global__ void init_state_kernel(float* __restrict__ hf, bf16_t* __restrict__ hb,
                                  unsigned* __restrict__ bar) {
  int i = blockIdx.x * 256 + threadIdx.x;            // over B*H
  if (i == 0) *bar = 0u;
  hf[i] = 0.0f;
  hb[i] = (bf16_t)0.0f;
}

// ---------------- big input-projection GEMM --------------------------------
// out_bf[M,N] = bf16( A_bf[M,K] @ W (via Wt[N,K]) + bias[N] ), 16x64 per wave.
// Register double-buffered: fragments for k+32 are in flight while the four
// WMMAs for k execute, so v_wmma no longer waits on its own loads.
__global__ void __launch_bounds__(256)
gemm_x_w_kernel(const bf16_t* __restrict__ A, const bf16_t* __restrict__ Bt,
                const float* __restrict__ bias, bf16_t* __restrict__ out,
                int M, int N, int K) {
  int w  = (blockIdx.x * 256 + threadIdx.x) >> 5;
  int n4 = N >> 6;
  int mt = w / n4;
  int n0 = (w % n4) << 6;
  const bf16_t* Abase = A + (long)mt * 16 * K;
  const bf16_t* Bbase[4];
#pragma unroll
  for (int j = 0; j < 4; ++j) Bbase[j] = Bt + (long)(n0 + j * 16) * K;

  v8f zf = {};
  v8f acc[4];
#pragma unroll
  for (int j = 0; j < 4; ++j) acc[j] = zf;

  v16bf aC = load_fragA(Abase, K);
  v16bf bC[4];
#pragma unroll
  for (int j = 0; j < 4; ++j) bC[j] = load_fragB(Bbase[j], K);

  for (int k0 = 0; k0 < K; k0 += 32) {
    const int kn = k0 + 32;
    v16bf aN;
    v16bf bN[4];
    if (kn < K) {
      __builtin_prefetch((const void*)(Abase + kn + 32), 0, 0);
      aN = load_fragA(Abase + kn, K);
#pragma unroll
      for (int j = 0; j < 4; ++j) bN[j] = load_fragB(Bbase[j] + kn, K);
    }
#pragma unroll
    for (int j = 0; j < 4; ++j) acc[j] = wmma_bf16(aC, bC[j], acc[j]);
    if (kn < K) {
      aC = aN;
#pragma unroll
      for (int j = 0; j < 4; ++j) bC[j] = bN[j];
    }
  }

  int lane = threadIdx.x & 31;
  int rofs = (lane >> 4) << 3;
  int c    = lane & 15;
#pragma unroll
  for (int j = 0; j < 4; ++j) {
    int col  = n0 + j * 16 + c;
    float bv = bias[col];
#pragma unroll
    for (int v = 0; v < 8; ++v) {
      int row = mt * 16 + v + rofs;
      out[(long)row * N + col] = (bf16_t)(acc[j][v] + bv);
    }
  }
}

// ---------------- persistent recurrent scan --------------------------------
// 32 blocks x 128 threads (4 waves). Per block LDS:
//   whzr_lds: 4 gate strips  (64 rows x 1024 K)  = 128 KB
//   whh_lds : 2 cand strips  (32 rows x 1024 K)  =  64 KB
// Gate strips:  global wave g = blockIdx*4+wib, cols [g*16, g*16+16) of 2H.
// Cand strips:  s = blockIdx*2+wib (wib<2),     cols [s*16, s*16+16) of H.
__global__ void __launch_bounds__(SCAN_THREADS)
gru_scan_kernel(const bf16_t* __restrict__ WhzrT,  // [2H, H] bf16
                const bf16_t* __restrict__ WhhT,   // [H, H]  bf16
                const float*  __restrict__ bhzr,   // [2H]
                const float*  __restrict__ bhh,    // [H]
                const bf16_t* __restrict__ gx,     // [L, B, 3H] bf16
                const float*  __restrict__ mask,   // [L, B]
                float* __restrict__ h_f, bf16_t* __restrict__ h_b,
                float* __restrict__ zbuf, bf16_t* __restrict__ rh,
                float* __restrict__ out_f, bf16_t* __restrict__ out_b,
                unsigned* __restrict__ bar, int reverse) {
  __shared__ __align__(32) bf16_t whzr_lds[4 * 16 * Hc];  // 128 KB
  __shared__ __align__(32) bf16_t whh_lds[2 * 16 * Hc];   //  64 KB

  const int tid   = threadIdx.x;
  const int blk   = blockIdx.x;
  const int wib   = tid >> 5;       // wave in block, 0..3
  const int lane  = tid & 31;
  const int rofs  = (lane >> 4) << 3;
  const int cidx  = lane & 15;

  // ---- one-time weight preload to LDS via async global->LDS copies ----
  {
    const bf16_t* src0 = WhzrT + (size_t)blk * 64 * Hc;   // 64 rows
    for (int i = tid; i < (4 * 16 * Hc) / 8; i += SCAN_THREADS) {
      unsigned lds_off = (unsigned)(unsigned long long)(const void*)&whzr_lds[i * 8];
      const bf16_t* g  = src0 + (size_t)i * 8;
      asm volatile("global_load_async_to_lds_b128 %0, %1, off"
                   :: "v"(lds_off), "v"(g) : "memory");
    }
    const bf16_t* src1 = WhhT + (size_t)blk * 32 * Hc;    // 32 rows
    for (int i = tid; i < (2 * 16 * Hc) / 8; i += SCAN_THREADS) {
      unsigned lds_off = (unsigned)(unsigned long long)(const void*)&whh_lds[i * 8];
      const bf16_t* g  = src1 + (size_t)i * 8;
      asm volatile("global_load_async_to_lds_b128 %0, %1, off"
                   :: "v"(lds_off), "v"(g) : "memory");
    }
    asm volatile("s_wait_asynccnt 0x0" ::: "memory");
    __syncthreads();
  }

  unsigned episode = 0;

  for (int step = 0; step < Lc; ++step) {
    const int t = reverse ? (Lc - 1 - step) : step;
    const bf16_t* gx_t = gx + (size_t)t * Bc * G3;

    // ================= phase 1: gates (all 128 waves) =================
    {
      const int n0 = (blk * 4 + wib) * 16;   // column strip of [0, 2H)
      v8f zf = {};
      v8f acc[4];
#pragma unroll
      for (int m = 0; m < 4; ++m) acc[m] = zf;

      v16bf aC[4], aN[4];
#pragma unroll
      for (int m = 0; m < 4; ++m) aC[m] = load_fragA(h_b + m * 16 * Hc, Hc);

      const int bbase = wib * 16 * Hc + cidx * Hc + ((lane >> 4) << 4);
#pragma unroll 2
      for (int k0 = 0; k0 < Hc; k0 += 32) {
        const bool more = (k0 + 32 < Hc);
        if (more) {
#pragma unroll
          for (int m = 0; m < 4; ++m)
            aN[m] = load_fragA(h_b + m * 16 * Hc + k0 + 32, Hc);
        }
        v16bf bfrag = *(const v16bf*)&whzr_lds[bbase + k0];
#pragma unroll
        for (int m = 0; m < 4; ++m) acc[m] = wmma_bf16(aC[m], bfrag, acc[m]);
        if (more) {
#pragma unroll
          for (int m = 0; m < 4; ++m) aC[m] = aN[m];
        }
      }

      const int col = n0 + cidx;
      const float bv = bhzr[col];
#pragma unroll
      for (int m = 0; m < 4; ++m) {
#pragma unroll
        for (int v = 0; v < 8; ++v) {
          int row = m * 16 + v + rofs;
          float g = acc[m][v] + (float)gx_t[(long)row * G3 + col] + bv;
          float s = sigmoidf_(g);
          if (col < Hc) {
            zbuf[row * Hc + col] = s;                       // z gate
          } else {
            int j = col - Hc;                               // r gate
            rh[row * Hc + j] = (bf16_t)(s * h_f[row * Hc + j]);
          }
        }
      }
    }
    grid_barrier(bar, (++episode) * SCAN_BLOCKS);

    // ================= phase 2: candidate + h update (64 waves) ========
    if (wib < 2) {
      const int n0 = (blk * 2 + wib) * 16;   // column strip of [0, H)
      v8f zf = {};
      v8f acc[4];
#pragma unroll
      for (int m = 0; m < 4; ++m) acc[m] = zf;

      v16bf aC[4], aN[4];
#pragma unroll
      for (int m = 0; m < 4; ++m) aC[m] = load_fragA(rh + m * 16 * Hc, Hc);

      const int bbase = wib * 16 * Hc + cidx * Hc + ((lane >> 4) << 4);
#pragma unroll 2
      for (int k0 = 0; k0 < Hc; k0 += 32) {
        const bool more = (k0 + 32 < Hc);
        if (more) {
#pragma unroll
          for (int m = 0; m < 4; ++m)
            aN[m] = load_fragA(rh + m * 16 * Hc + k0 + 32, Hc);
        }
        v16bf bfrag = *(const v16bf*)&whh_lds[bbase + k0];
#pragma unroll
        for (int m = 0; m < 4; ++m) acc[m] = wmma_bf16(aC[m], bfrag, acc[m]);
        if (more) {
#pragma unroll
          for (int m = 0; m < 4; ++m) aC[m] = aN[m];
        }
      }

      const int col = n0 + cidx;
      const float bv = bhh[col];
#pragma unroll
      for (int m = 0; m < 4; ++m) {
#pragma unroll
        for (int v = 0; v < 8; ++v) {
          int row = m * 16 + v + rofs;
          int idx = row * Hc + col;
          float hc = tanhf(acc[m][v] + (float)gx_t[(long)row * G3 + 2 * Hc + col] + bv);
          float zv = zbuf[idx];
          float ho = h_f[idx];
          float hn = (1.0f - zv) * ho + zv * hc;
          float mk = mask[t * Bc + row];
          hn = mk * hn + (1.0f - mk) * ho;
          h_f[idx]  = hn;
          h_b[idx]  = (bf16_t)hn;
          if (out_f) out_f[(size_t)t * Bc * Hc + idx] = hn;
          else       out_b[(size_t)t * Bc * Hc + idx] = (bf16_t)hn;
        }
      }
    }
    grid_barrier(bar, (++episode) * SCAN_BLOCKS);
  }
}

// ---------------------------------------------------------------------------
extern "C" void kernel_launch(void* const* d_in, const int* in_sizes, int n_in,
                              void* d_out, int out_size, void* d_ws, size_t ws_size,
                              hipStream_t stream) {
  (void)in_sizes; (void)n_in; (void)out_size; (void)ws_size;

  const int*   xs       = (const int*)  d_in[0];
  const float* xs_mask  = (const float*)d_in[1];
  const float* emb      = (const float*)d_in[2];
  const float* fw_Wx    = (const float*)d_in[3];
  const float* fw_bx    = (const float*)d_in[4];
  const float* fw_Whzr  = (const float*)d_in[5];
  const float* fw_bhzr  = (const float*)d_in[6];
  const float* fw_Whh   = (const float*)d_in[7];
  const float* fw_bhh   = (const float*)d_in[8];
  const float* bw_Wx    = (const float*)d_in[9];
  const float* bw_bx    = (const float*)d_in[10];
  const float* bw_Whzr  = (const float*)d_in[11];
  const float* bw_bhzr  = (const float*)d_in[12];
  const float* bw_Whh   = (const float*)d_in[13];
  const float* bw_bhh   = (const float*)d_in[14];
  float* out = (float*)d_out;

  // ---- workspace carve-up (256B aligned slices) ----
  char* wsb = (char*)d_ws;
  size_t cur = 0;
  auto take = [&](size_t bytes) -> char* {
    char* p = wsb + cur;
    cur += (bytes + 255) & ~(size_t)255;
    return p;
  };
  bf16_t* xe       = (bf16_t*)take((size_t)Lc * Bc * Ec * 2);      // 8 MB
  bf16_t* fwWxT    = (bf16_t*)take((size_t)G3 * Ec * 2);           // 3 MB
  bf16_t* fwWhzrT  = (bf16_t*)take((size_t)2 * Hc * Hc * 2);       // 4 MB
  bf16_t* fwWhhT   = (bf16_t*)take((size_t)Hc * Hc * 2);           // 2 MB
  bf16_t* bwWxT    = (bf16_t*)take((size_t)G3 * Hc * 2);           // 6 MB
  bf16_t* bwWhzrT  = (bf16_t*)take((size_t)2 * Hc * Hc * 2);       // 4 MB
  bf16_t* bwWhhT   = (bf16_t*)take((size_t)Hc * Hc * 2);           // 2 MB
  bf16_t* gx       = (bf16_t*)take((size_t)Lc * Bc * G3 * 2);      // 48 MB (reused)
  bf16_t* right    = (bf16_t*)take((size_t)Lc * Bc * Hc * 2);      // 16 MB
  float*  h_f      = (float*) take((size_t)Bc * Hc * 4);
  bf16_t* h_b      = (bf16_t*)take((size_t)Bc * Hc * 2);
  float*  zbuf     = (float*) take((size_t)Bc * Hc * 4);
  bf16_t* rh       = (bf16_t*)take((size_t)Bc * Hc * 2);
  unsigned* bar    = (unsigned*)take(256);

  // ---- 1) embedding gather -> bf16 ----
  embed_cvt_kernel<<<(Lc * Bc * Ec) / 256, 256, 0, stream>>>(xs, emb, xe);

  // ---- 2) weight transposes -> bf16 N-major/K-contiguous ----
  auto tgrid = [](long k, long n) { return (unsigned)((k * n + 255) / 256); };
  transpose_cvt_kernel<<<tgrid(Ec, G3),     256, 0, stream>>>(fw_Wx,   fwWxT,   Ec, G3);
  transpose_cvt_kernel<<<tgrid(Hc, 2 * Hc), 256, 0, stream>>>(fw_Whzr, fwWhzrT, Hc, 2 * Hc);
  transpose_cvt_kernel<<<tgrid(Hc, Hc),     256, 0, stream>>>(fw_Whh,  fwWhhT,  Hc, Hc);
  transpose_cvt_kernel<<<tgrid(Hc, G3),     256, 0, stream>>>(bw_Wx,   bwWxT,   Hc, G3);
  transpose_cvt_kernel<<<tgrid(Hc, 2 * Hc), 256, 0, stream>>>(bw_Whzr, bwWhzrT, Hc, 2 * Hc);
  transpose_cvt_kernel<<<tgrid(Hc, Hc),     256, 0, stream>>>(bw_Whh,  bwWhhT,  Hc, Hc);

  const int MR = Lc * Bc;  // 8192 GEMM rows
  const unsigned gemm_blocks = (unsigned)(((MR / 16) * (G3 / 64)) / 8); // 3072

  // ---- 3) forward input projection + forward scan ----
  gemm_x_w_kernel<<<gemm_blocks, 256, 0, stream>>>(xe, fwWxT, fw_bx, gx, MR, G3, Ec);
  init_state_kernel<<<(Bc * Hc) / 256, 256, 0, stream>>>(h_f, h_b, bar);
  gru_scan_kernel<<<SCAN_BLOCKS, SCAN_THREADS, 0, stream>>>(
      fwWhzrT, fwWhhT, fw_bhzr, fw_bhh, gx, xs_mask,
      h_f, h_b, zbuf, rh, /*out_f=*/nullptr, /*out_b=*/right, bar, /*reverse=*/0);

  // ---- 4) backward input projection + backward scan ----
  gemm_x_w_kernel<<<gemm_blocks, 256, 0, stream>>>(right, bwWxT, bw_bx, gx, MR, G3, Hc);
  init_state_kernel<<<(Bc * Hc) / 256, 256, 0, stream>>>(h_f, h_b, bar);
  gru_scan_kernel<<<SCAN_BLOCKS, SCAN_THREADS, 0, stream>>>(
      bwWhzrT, bwWhhT, bw_bhzr, bw_bhh, gx, xs_mask,
      h_f, h_b, zbuf, rh, /*out_f=*/out, /*out_b=*/nullptr, bar, /*reverse=*/1);
}
